// GlobalFluxTracker_18356690223184
// MI455X (gfx1250) — matrix operand order repaired and checked by reading, not compiled, not run on hardware
//
#include <hip/hip_runtime.h>
#include <math.h>

// MI455X / gfx1250, wave32. bf16 WMMA (v_wmma_f32_16x16x32_bf16) GEMMs with
// f32 accumulate; 320KB-WGP-LDS-resident weight slabs; chunked parallel scan;
// GEMM2 uses double-buffered GLOBAL_LOAD_ASYNC_TO_LDS_B128 (ASYNCcnt) copies.

typedef __attribute__((ext_vector_type(8)))  __bf16 v8bf;
typedef __attribute__((ext_vector_type(16))) __bf16 v16bf;
typedef __attribute__((ext_vector_type(8)))  float  v8f;
typedef __attribute__((ext_vector_type(4)))  int    v4i;

namespace gft {
constexpr int kB = 8, kT = 4096, kD = 256;
constexpr int kK = 2 * kD;            // 512  (GEMM K dim)
constexpr int kM = kB * kT;           // 32768 (rows = B*T)
constexpr int kN2 = 3 * kD;           // 768  (source re|im + gate)
constexpr int MT = 128, NT = 128, KT = 32;
constexpr int WP = kK + 8;            // LDS weight pitch (bank-conflict-free)
constexpr int AP = KT + 8;            // LDS A pitch
constexpr size_t WLDS_BYTES = (size_t)NT * WP * sizeof(__bf16);  // 133120
constexpr size_t ALDS_BYTES = (size_t)MT * AP * sizeof(__bf16);  // 10240
constexpr size_t SMEM_BYTES = WLDS_BYTES + 2 * ALDS_BYTES;       // ~150KB < 320KB WGP LDS
constexpr int CHUNK = 64, NCHUNK = kT / CHUNK;                   // 64 chunks of 64
}  // namespace gft

using namespace gft;

__device__ __forceinline__ float sigmoidf_(float x) {
  return 1.0f / (1.0f + __expf(-x));
}

// ---- CDNA5 async global->LDS copy (CDNA5 ISA) --------
typedef __attribute__((address_space(1))) v4i* gv4i_p;
typedef __attribute__((address_space(3))) v4i* lv4i_p;

__device__ __forceinline__ void async_copy_b128(const void* gsrc, void* ldst) {
#if __has_builtin(__builtin_amdgcn_global_load_async_to_lds_b128)
  __builtin_amdgcn_global_load_async_to_lds_b128(
      (gv4i_p)gsrc, (lv4i_p)ldst, 0, 0);
#else
  // VGLOBAL GV mode: vdst = LDS byte offset, v[addr] = 64-bit global address.
  unsigned loff = (unsigned)(unsigned long long)
      (__attribute__((address_space(3))) char*)(ldst);
  unsigned long long ga = (unsigned long long)gsrc;
  asm volatile("global_load_async_to_lds_b128 %0, %1, off"
               :: "v"(loff), "v"(ga) : "memory");
#endif
}

__device__ __forceinline__ void wait_async0() {
#if __has_builtin(__builtin_amdgcn_s_wait_asynccnt)
  __builtin_amdgcn_s_wait_asynccnt(0);
#else
  asm volatile("s_wait_asynccnt 0x0" ::: "memory");
#endif
}

// ---- WMMA fragment loads (CDNA5 16-bit layouts, CDNA5 ISA) -------
// A 16x32 bf16: lanes 0-15 (M=lane): K0..7 in v0..3, K16..23 in v4..7;
//               lanes 16-31 (M=lane-16): K8..15 and K24..31.
__device__ __forceinline__ v16bf load_afrag(const __bf16* Alds, int row, int lane) {
  const int koff = (lane >> 4) << 3;  // 0 or 8
  const v8bf lo = *(const v8bf*)(Alds + row * AP + koff);
  const v8bf hi = *(const v8bf*)(Alds + row * AP + 16 + koff);
  return __builtin_shufflevector(lo, hi, 0, 1, 2, 3, 4, 5, 6, 7,
                                 8, 9, 10, 11, 12, 13, 14, 15);
}
// B 32x16 bf16: lane n<16 holds column n, K0..7 and K16..23; lanes 16-31 K8..15 and K24..31.
// Weights stored [n][k] so this is two 16B loads per lane (NT gemm).
__device__ __forceinline__ v16bf load_bfrag(const __bf16* Wlds, int col, int kb, int lane) {
  const int koff = (lane >> 4) << 3;  // 0 or 8
  const v8bf lo = *(const v8bf*)(Wlds + col * WP + kb + koff);
  const v8bf hi = *(const v8bf*)(Wlds + col * WP + kb + koff + 16);
  return __builtin_shufflevector(lo, hi, 0, 1, 2, 3, 4, 5, 6, 7,
                                 8, 9, 10, 11, 12, 13, 14, 15);
}

// One K-step for a wave's 64x32 sub-tile: 4 m-frags x 2 n-frags -> 8 WMMAs.
__device__ __forceinline__ void wmma_step(const __bf16* Wlds, const __bf16* Alds,
                                          int kb, int wm, int wn, int lane,
                                          v8f acc[4][2]) {
  v16bf af[4];
#pragma unroll
  for (int mi = 0; mi < 4; ++mi)
    af[mi] = load_afrag(Alds, wm * 64 + mi * 16 + (lane & 15), lane);
  v16bf bq[2];
#pragma unroll
  for (int ni = 0; ni < 2; ++ni)
    bq[ni] = load_bfrag(Wlds, wn * 32 + ni * 16 + (lane & 15), kb, lane);
#pragma unroll
  for (int mi = 0; mi < 4; ++mi)
#pragma unroll
    for (int ni = 0; ni < 2; ++ni)
      acc[mi][ni] = __builtin_amdgcn_wmma_f32_16x16x32_bf16(
          false, af[mi], false, bq[ni], (short)0, acc[mi][ni], false, false);
}

// ---- GEMM1: U[m,n] = bm[n] + sum_k Xcat[m,k]*Wm[n,k]  (Xcat = [x_re|x_im]) --
__global__ void gft_gemm1(const float* __restrict__ x_re, const float* __restrict__ x_im,
                          const float* __restrict__ Wm, const float* __restrict__ bm,
                          float* __restrict__ U) {
  extern __shared__ char smem[];
  __bf16* Wlds = (__bf16*)smem;
  __bf16* Alds = (__bf16*)(smem + WLDS_BYTES);

  const int lane = threadIdx.x & 31;
  const int wid = threadIdx.x >> 5;
  const int wm = wid & 1, wn = wid >> 1;
  const int nbase = blockIdx.y * NT;

  // Load weight slab once per block: 128 n-rows x 512 k, f32 -> bf16.
  for (int idx = threadIdx.x; idx < NT * kK / 4; idx += blockDim.x) {
    const int n = idx >> 7;             // kK/4 = 128 float4 per row
    const int k4 = (idx & 127) << 2;
    const float4 v = *(const float4*)(Wm + (size_t)(nbase + n) * kK + k4);
    __bf16* d = Wlds + n * WP + k4;
    d[0] = (__bf16)v.x; d[1] = (__bf16)v.y; d[2] = (__bf16)v.z; d[3] = (__bf16)v.w;
  }
  __syncthreads();

  for (int mt = blockIdx.x; mt < kM / MT; mt += gridDim.x) {
    const int mbase = mt * MT;
    v8f acc[4][2];
#pragma unroll
    for (int mi = 0; mi < 4; ++mi)
#pragma unroll
      for (int ni = 0; ni < 2; ++ni)
        acc[mi][ni] = v8f{0.f, 0.f, 0.f, 0.f, 0.f, 0.f, 0.f, 0.f};

    for (int kb = 0; kb < kK; kb += KT) {
      __syncthreads();
      const float* Asrc = (kb < kD) ? x_re : x_im;
      const int kcol = (kb < kD) ? kb : kb - kD;
      for (int idx = threadIdx.x; idx < MT * KT / 4; idx += blockDim.x) {
        const int r = idx >> 3;
        const int c4 = (idx & 7) << 2;
        const float* src = Asrc + (size_t)(mbase + r) * kD + kcol + c4;
        __builtin_prefetch(src + (size_t)MT * kD, 0, 0);
        const float4 v = *(const float4*)src;
        __bf16* d = Alds + r * AP + c4;
        d[0] = (__bf16)v.x; d[1] = (__bf16)v.y; d[2] = (__bf16)v.z; d[3] = (__bf16)v.w;
      }
      __syncthreads();
      wmma_step(Wlds, Alds, kb, wm, wn, lane, acc);
    }

    // Epilogue: C/D layout — VGPR r: lanes 0-15 -> M=r, lanes 16-31 -> M=r+8.
    // Single base pointer per accumulator; r*kK folds into store imm offset.
    const int half8 = (lane >> 4) << 3;
    const int lcol = lane & 15;
#pragma unroll
    for (int mi = 0; mi < 4; ++mi)
#pragma unroll
      for (int ni = 0; ni < 2; ++ni) {
        const int ng = nbase + wn * 32 + ni * 16 + lcol;
        const float bias = bm[ng];
        float* p = U + (size_t)(mbase + wm * 64 + mi * 16 + half8) * kK + ng;
#pragma unroll
        for (int r = 0; r < 8; ++r) p[r * kK] = acc[mi][ni][r] + bias;
      }
  }
}

// ---- Scan phase 1: per-(b,chunk,d) partial P = sum decay^(L-1-j) * u_j -----
__global__ void gft_scan1(const float* __restrict__ U,
                          const float* __restrict__ decay_re,
                          const float* __restrict__ decay_im,
                          float* __restrict__ Pre, float* __restrict__ Pim) {
  const int tid = blockIdx.x * blockDim.x + threadIdx.x;  // 131072
  const int d = tid & 255;
  const int c = (tid >> 8) & (NCHUNK - 1);
  const int b = tid >> 14;
  const float dr = sigmoidf_(decay_re[d]);
  const float di = decay_im[d];
  float pr = 0.f, pi = 0.f;
  size_t base = ((size_t)b * kT + (size_t)c * CHUNK) * kK + d;
#pragma unroll 4
  for (int j = 0; j < CHUNK; ++j) {
    __builtin_prefetch(&U[base + 8 * kK], 0, 0);
    const float ur = U[base];
    const float ui = U[base + kD];
    const float nr = pr * dr - pi * di + ur;
    const float ni = pr * di + pi * dr + ui;
    pr = nr; pi = ni;
    base += kK;
  }
  const int idx = ((b << 6) + c) * kD + d;
  Pre[idx] = pr; Pim[idx] = pi;
}

// ---- Scan phase 2: per-channel scan over 64 chunks with decay^64 -----------
__global__ void gft_scan2(const float* __restrict__ s0_re, const float* __restrict__ s0_im,
                          const float* __restrict__ decay_re, const float* __restrict__ decay_im,
                          const float* __restrict__ Pre, const float* __restrict__ Pim,
                          float* __restrict__ SinRe, float* __restrict__ SinIm) {
  const int tid = blockIdx.x * blockDim.x + threadIdx.x;  // 2048
  const int d = tid & 255;
  const int b = tid >> 8;
  const float dr = sigmoidf_(decay_re[d]);
  const float di = decay_im[d];
  float ar = dr, ai = di;            // decay^64 by 6 squarings
#pragma unroll
  for (int i = 0; i < 6; ++i) {
    const float nr = ar * ar - ai * ai;
    const float ni = 2.f * ar * ai;
    ar = nr; ai = ni;
  }
  float sr = s0_re[tid], si = s0_im[tid];
  for (int c = 0; c < NCHUNK; ++c) {
    const int idx = ((b << 6) + c) * kD + d;
    SinRe[idx] = sr; SinIm[idx] = si;
    const float nr = sr * ar - si * ai + Pre[idx];
    const float ni = sr * ai + si * ar + Pim[idx];
    sr = nr; si = ni;
  }
}

// ---- Scan phase 3: replay chunks with correct incoming state, emit bf16 ----
__global__ void gft_scan3(const float* __restrict__ U,
                          const float* __restrict__ decay_re,
                          const float* __restrict__ decay_im,
                          const float* __restrict__ SinRe, const float* __restrict__ SinIm,
                          __bf16* __restrict__ Sbf) {
  const int tid = blockIdx.x * blockDim.x + threadIdx.x;
  const int d = tid & 255;
  const int c = (tid >> 8) & (NCHUNK - 1);
  const int b = tid >> 14;
  const float dr = sigmoidf_(decay_re[d]);
  const float di = decay_im[d];
  const int idx = ((b << 6) + c) * kD + d;
  float sr = SinRe[idx], si = SinIm[idx];
  size_t base = ((size_t)b * kT + (size_t)c * CHUNK) * kK + d;
#pragma unroll 4
  for (int j = 0; j < CHUNK; ++j) {
    __builtin_prefetch(&U[base + 8 * kK], 0, 0);
    const float ur = U[base];
    const float ui = U[base + kD];
    const float nr = sr * dr - si * di + ur;
    const float ni = sr * di + si * dr + ui;
    sr = nr; si = ni;
    Sbf[base] = (__bf16)nr;
    Sbf[base + kD] = (__bf16)ni;
    base += kK;
  }
}

// ---- GEMM2: out[m, 0:512] = Scat@Wo.T + bo ; out[m,512:768] = clip(sig(Scat@Wg.T+bg))
// A tiles (already bf16) streamed with double-buffered async global->LDS DMA.
__global__ void gft_gemm2(const __bf16* __restrict__ Sbf,
                          const float* __restrict__ Wo, const float* __restrict__ Wg,
                          const float* __restrict__ bo, const float* __restrict__ bg,
                          float* __restrict__ out) {
  extern __shared__ char smem[];
  __bf16* Wlds = (__bf16*)smem;
  __bf16* A0 = (__bf16*)(smem + WLDS_BYTES);
  __bf16* A1 = A0 + MT * AP;

  const int lane = threadIdx.x & 31;
  const int wid = threadIdx.x >> 5;
  const int wm = wid & 1, wn = wid >> 1;
  const int nbase = blockIdx.y * NT;  // 0..640 (6 blocks) over [Wo rows | Wg rows]

  for (int idx = threadIdx.x; idx < NT * kK / 4; idx += blockDim.x) {
    const int n = idx >> 7;
    const int k4 = (idx & 127) << 2;
    const int ng = nbase + n;
    const float* src = (ng < kK) ? (Wo + (size_t)ng * kK + k4)
                                 : (Wg + (size_t)(ng - kK) * kK + k4);
    const float4 v = *(const float4*)src;
    __bf16* d = Wlds + n * WP + k4;
    d[0] = (__bf16)v.x; d[1] = (__bf16)v.y; d[2] = (__bf16)v.z; d[3] = (__bf16)v.w;
  }
  __syncthreads();

  for (int mt = blockIdx.x; mt < kM / MT; mt += gridDim.x) {
    const int mbase = mt * MT;
    v8f acc[4][2];
#pragma unroll
    for (int mi = 0; mi < 4; ++mi)
#pragma unroll
      for (int ni = 0; ni < 2; ++ni)
        acc[mi][ni] = v8f{0.f, 0.f, 0.f, 0.f, 0.f, 0.f, 0.f, 0.f};

    // 2 x b128 async copies per thread per tile (128x32 bf16 = 8KB).
    auto issue_tile = [&](int kb, __bf16* buf) {
      for (int idx = threadIdx.x; idx < MT * KT / 8; idx += blockDim.x) {
        const int r = idx >> 2;
        const int c8 = (idx & 3) << 3;
        async_copy_b128(Sbf + (size_t)(mbase + r) * kK + kb + c8,
                        buf + r * AP + c8);
      }
    };

    issue_tile(0, A0);
    for (int kt = 0; kt < kK / KT; ++kt) {
      const int kb = kt * KT;
      wait_async0();        // this wave's async copies of tile kt done
      __syncthreads();      // everyone's copies visible; prev buffer now dead
      if (kt + 1 < kK / KT) issue_tile(kb + KT, (kt & 1) ? A0 : A1);
      wmma_step(Wlds, (kt & 1) ? A1 : A0, kb, wm, wn, lane, acc);
    }

    const int half8 = (lane >> 4) << 3;
    const int lcol = lane & 15;
#pragma unroll
    for (int mi = 0; mi < 4; ++mi)
#pragma unroll
      for (int ni = 0; ni < 2; ++ni) {
        const int ng = nbase + wn * 32 + ni * 16 + lcol;
        const bool is_gate = (ng >= kK);
        const float bias = is_gate ? bg[ng - kK] : bo[ng];
        float* p = out + (size_t)(mbase + wm * 64 + mi * 16 + half8) * kN2 + ng;
#pragma unroll
        for (int r = 0; r < 8; ++r) {
          float val = acc[mi][ni][r] + bias;
          if (is_gate) {
            val = sigmoidf_(val);
            val = fminf(0.99f, fmaxf(0.01f, val));
          }
          p[r * kN2] = val;
        }
      }
  }
}

extern "C" void kernel_launch(void* const* d_in, const int* in_sizes, int n_in,
                              void* d_out, int out_size, void* d_ws, size_t ws_size,
                              hipStream_t stream) {
  const float* x_re = (const float*)d_in[0];
  const float* x_im = (const float*)d_in[1];
  const float* s0_re = (const float*)d_in[2];
  const float* s0_im = (const float*)d_in[3];
  const float* decay_re = (const float*)d_in[4];
  const float* decay_im = (const float*)d_in[5];
  const float* Wm = (const float*)d_in[6];
  const float* bm = (const float*)d_in[7];
  const float* Wo = (const float*)d_in[8];
  const float* bo = (const float*)d_in[9];
  const float* Wg = (const float*)d_in[10];
  const float* bg = (const float*)d_in[11];
  float* out = (float*)d_out;

  // Workspace layout (bytes):
  //   U (f32, M x 512)    : [0,           67108864)
  //   Sbf (bf16, M x 512) : [67108864,    100663296)
  //   Pre/Pim/SinRe/SinIm : 4 x 512KB after that  (~98 MB total)
  char* ws = (char*)d_ws;
  float* U = (float*)ws;
  __bf16* Sbf = (__bf16*)(ws + 67108864u);
  float* Pre = (float*)(ws + 100663296u);
  float* Pim = Pre + kB * NCHUNK * kD;
  float* SinRe = Pim + kB * NCHUNK * kD;
  float* SinIm = SinRe + kB * NCHUNK * kD;

  // ~150KB dynamic LDS (WGP has 320KB); opt in above the 64KB default cap.
  (void)hipFuncSetAttribute(reinterpret_cast<const void*>(gft_gemm1),
                            hipFuncAttributeMaxDynamicSharedMemorySize, (int)SMEM_BYTES);
  (void)hipFuncSetAttribute(reinterpret_cast<const void*>(gft_gemm2),
                            hipFuncAttributeMaxDynamicSharedMemorySize, (int)SMEM_BYTES);

  gft_gemm1<<<dim3(64, 4), 256, SMEM_BYTES, stream>>>(x_re, x_im, Wm, bm, U);
  gft_scan1<<<512, 256, 0, stream>>>(U, decay_re, decay_im, Pre, Pim);
  gft_scan2<<<8, 256, 0, stream>>>(s0_re, s0_im, decay_re, decay_im, Pre, Pim, SinRe, SinIm);
  gft_scan3<<<512, 256, 0, stream>>>(U, decay_re, decay_im, SinRe, SinIm, Sbf);
  gft_gemm2<<<dim3(64, 6), 256, SMEM_BYTES, stream>>>(Sbf, Wo, Wg, bo, bg, out);
}
